// MoeLayer_14869176779218
// MI455X (gfx1250) — matrix operand order repaired and checked by reading, not compiled, not run on hardware
//
#include <hip/hip_runtime.h>

#define T_TOK 4096
#define DIM   2048
#define FDIM  4096
#define NEXP  8

typedef __attribute__((ext_vector_type(16))) __bf16          v16bf;
typedef __attribute__((ext_vector_type(8)))  float           v8f;
typedef __attribute__((ext_vector_type(8)))  unsigned short  ushort8;
typedef __attribute__((ext_vector_type(4)))  float           float4v;

// ---- optional gfx1250 async global->LDS path (guarded; falls back cleanly) ----
#if defined(__has_builtin)
#if __has_builtin(__builtin_amdgcn_global_load_async_to_lds_b128) && \
    __has_builtin(__builtin_amdgcn_s_wait_asynccnt)
#define USE_ASYNC_LDS 1
#endif
#endif
#ifndef USE_ASYNC_LDS
#define USE_ASYNC_LDS 0
#endif

#if USE_ASYNC_LDS
// The builtin's pointer parameters are vector-typed (int32x4* for b128) and may
// carry address-space qualifiers. These adapters deduce the exact parameter
// type via a templated conversion operator; the C-style cast inside performs
// the pointee + address-space conversion (the form clang accepts on amdgcn).
struct GblPtrAdapter {
    const void* p;
    template <typename T>
    __device__ __forceinline__ operator T*() const { return (T*)(void*)p; }
};
struct LdsPtrAdapter {
    void* p;
    template <typename T>
    __device__ __forceinline__ operator T*() const { return (T*)p; }
};
__device__ __forceinline__ void async_copy_b128(const void* g, void* l) {
    __builtin_amdgcn_global_load_async_to_lds_b128(
        GblPtrAdapter{g}, LdsPtrAdapter{l}, 0, 0);
}
#endif

__device__ __forceinline__ unsigned short f2bf(float f) {
    unsigned int u = __builtin_bit_cast(unsigned int, f);
    u += 0x7FFFu + ((u >> 16) & 1u);          // round-to-nearest-even
    return (unsigned short)(u >> 16);
}

// A fragment (16x32 MxK, 16-bit): lane r=lane&15 row M=r; half=lane>>4 selects
// K = {half*8 + 0..7, 16 + half*8 + 0..7}
__device__ __forceinline__ v16bf load_frag_A(const unsigned short (*s)[32], int row, int half) {
    union { v16bf v; ushort8 h[2]; } f;
    f.h[0] = *(const ushort8*)&s[row][half * 8];
    f.h[1] = *(const ushort8*)&s[row][16 + half * 8];
    return f.v;
}

// B fragment (32x16 KxN, 16-bit): half=lane>>4 holds K = {half*8 + 0..7, 16 + half*8 + 0..7};
// LDS tile stored transposed as [n][k] so this is two contiguous b128 reads.
__device__ __forceinline__ v16bf load_frag_B(const unsigned short (*s)[32], int row, int half) {
    union { v16bf v; ushort8 h[2]; } f;
    f.h[0] = *(const ushort8*)&s[row][half * 8];
    f.h[1] = *(const ushort8*)&s[row][16 + half * 8];
    return f.v;
}

// ---------------------------------------------------------------------------
// Kernel 1: gating. One wave per token: logits = x[t] @ gate_w, top-2, softmax,
// append assignment (t*2 + k) to the owning expert's list.
// ---------------------------------------------------------------------------
__global__ void __launch_bounds__(256) gate_kernel(
        const float* __restrict__ x,
        const float* __restrict__ gw,
        float* __restrict__ topw,      // [T,2] flat: topw[t*2+k]
        int*   __restrict__ cnt,       // [E], pre-zeroed
        int*   __restrict__ lists)     // [E*T]
{
    const int tok  = (int)((blockIdx.x * blockDim.x + threadIdx.x) >> 5);
    const int lane = threadIdx.x & 31;
    if (tok >= T_TOK) return;

    const float* xr = x + (size_t)tok * DIM;
    float acc[NEXP];
#pragma unroll
    for (int e = 0; e < NEXP; ++e) acc[e] = 0.f;

    for (int d = lane; d < DIM; d += 32) {
        float xv = xr[d];
        const float* g = gw + (size_t)d * NEXP;
#pragma unroll
        for (int e = 0; e < NEXP; ++e) acc[e] += xv * g[e];
    }
#pragma unroll
    for (int e = 0; e < NEXP; ++e) {
#pragma unroll
        for (int off = 16; off > 0; off >>= 1)
            acc[e] += __shfl_xor(acc[e], off, 32);
    }

    if (lane == 0) {
        int b0 = 0; float v0 = acc[0];
#pragma unroll
        for (int e = 1; e < NEXP; ++e) if (acc[e] > v0) { v0 = acc[e]; b0 = e; }
        int b1 = (b0 == 0) ? 1 : 0; float v1 = acc[b1];
#pragma unroll
        for (int e = 0; e < NEXP; ++e)
            if (e != b0 && acc[e] > v1) { v1 = acc[e]; b1 = e; }

        float e0 = 1.f;                    // exp(v0 - v0)
        float e1 = __expf(v1 - v0);
        float inv = 1.f / (e0 + e1);
        topw[tok * 2 + 0] = e0 * inv;
        topw[tok * 2 + 1] = e1 * inv;

        int p0 = atomicAdd(&cnt[b0], 1);
        lists[b0 * T_TOK + p0] = tok * 2 + 0;
        int p1 = atomicAdd(&cnt[b1], 1);
        lists[b1 * T_TOK + p1] = tok * 2 + 1;
    }
}

// ---------------------------------------------------------------------------
// Kernel 2: x fp32 -> bf16 (x is re-read ~64x in FFN-up; bf16 copy is L2-resident)
// ---------------------------------------------------------------------------
__global__ void __launch_bounds__(256) cvt_x_kernel(
        const float* __restrict__ x, unsigned short* __restrict__ xb)
{
    size_t i = ((size_t)blockIdx.x * blockDim.x + threadIdx.x) * 8;
    float4v a = *(const float4v*)(x + i);
    float4v b = *(const float4v*)(x + i + 4);
    ushort8 o;
#pragma unroll
    for (int j = 0; j < 4; ++j) { o[j] = f2bf(a[j]); o[4 + j] = f2bf(b[j]); }
    *(ushort8*)(xb + i) = o;
}

// ---------------------------------------------------------------------------
// Kernel 3: FFN-up. Per (expert, 64-token tile, 64-F tile): h = Xe@W1, g = Xe@W3,
// act = silu(h)*g stored bf16 at row = assignment slot (t*2+k).
// Double-buffered LDS: one barrier per K-step.
// ---------------------------------------------------------------------------
__global__ void __launch_bounds__(256) ffn_up_kernel(
        const unsigned short* __restrict__ xb,
        const float* __restrict__ w1,
        const float* __restrict__ w3,
        const int*   __restrict__ cnt,
        const int*   __restrict__ lists,
        unsigned short* __restrict__ act)
{
    __shared__ unsigned short sA [2][64][32];   // [m][k] bf16 tokens
    __shared__ unsigned short sB1[2][64][32];   // [n][k] bf16 w1 (transposed)
    __shared__ unsigned short sB3[2][64][32];   // [n][k] bf16 w3 (transposed)
    __shared__ int sTok[64];

    const int e  = blockIdx.z;
    const int Ne = cnt[e];
    const int m0 = blockIdx.y * 64;
    if (m0 >= Ne) return;                    // block-uniform: EXEC stays full
    const int n0 = blockIdx.x * 64;

    const int tid = threadIdx.x;
    if (tid < 64) {
        int i = m0 + tid;
        sTok[tid] = (i < Ne) ? lists[e * T_TOK + i] : -1;
    }
    __syncthreads();

    // A-stage mapping: 64 rows x 32 cols, 4 threads/row, 8 bf16 (16B) each
    const int am = tid >> 2;
    const int ak = (tid & 3) * 8;
    const int aent = sTok[am];
    const size_t arow = (size_t)((aent < 0) ? 0 : (aent >> 1)) * DIM;

    // B-stage mapping: 32 k-rows x 64 n-cols, 8 threads/row, 8 fp32 each
    const int bk = tid >> 3;
    const int bn = (tid & 7) * 8;
    const float* W1 = w1 + (size_t)e * DIM * FDIM;
    const float* W3 = w3 + (size_t)e * DIM * FDIM;

    const int lane = tid & 31;
    const int wv   = tid >> 5;
    const int mi   = (wv & 3) * 16;          // 4 M-subtiles of 16
    const int nj   = (wv >> 2) * 32;         // 2 N-subtiles of 32
    const int r    = lane & 15;
    const int half = lane >> 4;

    v8f ch0 = {}; v8f ch1 = {}; v8f cg0 = {}; v8f cg1 = {};

    for (int kt = 0; kt < DIM / 32; ++kt) {
        const int k0  = kt * 32;
        const int buf = kt & 1;

        // stage A (raw bf16 copy -> async global->LDS when available)
#if USE_ASYNC_LDS
        async_copy_b128(xb + arow + k0 + ak, &sA[buf][am][ak]);
#else
        *(ushort8*)&sA[buf][am][ak] = *(const ushort8*)(xb + arow + k0 + ak);
#endif

        // stage B1/B3 (fp32 -> bf16 convert, transposed to [n][k])
        const float* p1 = W1 + (size_t)(k0 + bk) * FDIM + n0 + bn;
        const float* p3 = W3 + (size_t)(k0 + bk) * FDIM + n0 + bn;
        float4v u1a = *(const float4v*)p1;
        float4v u1b = *(const float4v*)(p1 + 4);
        float4v u3a = *(const float4v*)p3;
        float4v u3b = *(const float4v*)(p3 + 4);
#pragma unroll
        for (int j = 0; j < 4; ++j) {
            sB1[buf][bn + j][bk]     = f2bf(u1a[j]);
            sB1[buf][bn + 4 + j][bk] = f2bf(u1b[j]);
            sB3[buf][bn + j][bk]     = f2bf(u3a[j]);
            sB3[buf][bn + 4 + j][bk] = f2bf(u3b[j]);
        }
        if (k0 + 32 < DIM) {                 // global_prefetch_b8 next K-tile
            __builtin_prefetch(p1 + (size_t)32 * FDIM, 0, 1);
            __builtin_prefetch(p3 + (size_t)32 * FDIM, 0, 1);
        }
#if USE_ASYNC_LDS
        __builtin_amdgcn_s_wait_asynccnt(0);  // LDS writes from async path visible
#endif
        __syncthreads();

        v16bf a   = load_frag_A(sA[buf],  mi + r,      half);
        v16bf b10 = load_frag_B(sB1[buf], nj + r,      half);
        v16bf b11 = load_frag_B(sB1[buf], nj + 16 + r, half);
        v16bf b30 = load_frag_B(sB3[buf], nj + r,      half);
        v16bf b31 = load_frag_B(sB3[buf], nj + 16 + r, half);
        ch0 = __builtin_amdgcn_wmma_f32_16x16x32_bf16(false, a, false, b10, (short)0, ch0, false, false);
        ch1 = __builtin_amdgcn_wmma_f32_16x16x32_bf16(false, a, false, b11, (short)0, ch1, false, false);
        cg0 = __builtin_amdgcn_wmma_f32_16x16x32_bf16(false, a, false, b30, (short)0, cg0, false, false);
        cg1 = __builtin_amdgcn_wmma_f32_16x16x32_bf16(false, a, false, b31, (short)0, cg1, false, false);
        // no trailing barrier: ping-pong buffer + per-wave dscnt wait before the
        // next signal makes the WAR distance two barriers.
    }

    // epilogue: silu(h) * g  -> bf16 act[slot][n]
#pragma unroll
    for (int i = 0; i < 8; ++i) {
        int m   = mi + half * 8 + i;         // C layout: m = 8*half + vgpr
        int ent = sTok[m];
        if (ent < 0) continue;
        size_t rowo = (size_t)ent * FDIM + n0;
        float h0 = ch0[i], g0 = cg0[i];
        float h1 = ch1[i], g1 = cg1[i];
        float a0 = (h0 / (1.f + __expf(-h0))) * g0;
        float a1 = (h1 / (1.f + __expf(-h1))) * g1;
        act[rowo + nj + r]      = f2bf(a0);
        act[rowo + nj + 16 + r] = f2bf(a1);
    }
}

// ---------------------------------------------------------------------------
// Kernel 4: FFN-down. y = act @ W2[e], scaled by gate weight, atomically
// accumulated into out[t]. Each out element receives exactly 2 adds
// (commutative), so result is deterministic.
// ---------------------------------------------------------------------------
__global__ void __launch_bounds__(256) ffn_down_kernel(
        const unsigned short* __restrict__ act,
        const float* __restrict__ w2,
        const int*   __restrict__ cnt,
        const int*   __restrict__ lists,
        const float* __restrict__ topw,
        float* __restrict__ out)
{
    __shared__ unsigned short sA[2][64][32];
    __shared__ unsigned short sB[2][64][32];
    __shared__ int   sTok[64];
    __shared__ float sW[64];

    const int e  = blockIdx.z;
    const int Ne = cnt[e];
    const int m0 = blockIdx.y * 64;
    if (m0 >= Ne) return;
    const int n0 = blockIdx.x * 64;

    const int tid = threadIdx.x;
    if (tid < 64) {
        int i = m0 + tid;
        int ent = (i < Ne) ? lists[e * T_TOK + i] : -1;
        sTok[tid] = ent;
        sW[tid]   = (ent >= 0) ? topw[ent] : 0.f;
    }
    __syncthreads();

    const int am = tid >> 2;
    const int ak = (tid & 3) * 8;
    const int aent = sTok[am];
    const size_t arow = (size_t)((aent < 0) ? 0 : aent) * FDIM;   // act row = slot

    const int bk = tid >> 3;
    const int bn = (tid & 7) * 8;
    const float* W2 = w2 + (size_t)e * FDIM * DIM;

    const int lane = tid & 31;
    const int wv   = tid >> 5;
    const int mi   = (wv & 3) * 16;
    const int nj   = (wv >> 2) * 32;
    const int r    = lane & 15;
    const int half = lane >> 4;

    v8f c0 = {}; v8f c1 = {};

    for (int kt = 0; kt < FDIM / 32; ++kt) {
        const int k0  = kt * 32;
        const int buf = kt & 1;

#if USE_ASYNC_LDS
        async_copy_b128(act + arow + k0 + ak, &sA[buf][am][ak]);
#else
        *(ushort8*)&sA[buf][am][ak] = *(const ushort8*)(act + arow + k0 + ak);
#endif

        const float* p2 = W2 + (size_t)(k0 + bk) * DIM + n0 + bn;
        float4v ua = *(const float4v*)p2;
        float4v ub = *(const float4v*)(p2 + 4);
#pragma unroll
        for (int j = 0; j < 4; ++j) {
            sB[buf][bn + j][bk]     = f2bf(ua[j]);
            sB[buf][bn + 4 + j][bk] = f2bf(ub[j]);
        }
        if (k0 + 32 < FDIM) __builtin_prefetch(p2 + (size_t)32 * DIM, 0, 1);
#if USE_ASYNC_LDS
        __builtin_amdgcn_s_wait_asynccnt(0);
#endif
        __syncthreads();

        v16bf a  = load_frag_A(sA[buf], mi + r,      half);
        v16bf b0 = load_frag_B(sB[buf], nj + r,      half);
        v16bf b1 = load_frag_B(sB[buf], nj + 16 + r, half);
        c0 = __builtin_amdgcn_wmma_f32_16x16x32_bf16(false, a, false, b0, (short)0, c0, false, false);
        c1 = __builtin_amdgcn_wmma_f32_16x16x32_bf16(false, a, false, b1, (short)0, c1, false, false);
    }

#pragma unroll
    for (int i = 0; i < 8; ++i) {
        int m   = mi + half * 8 + i;
        int ent = sTok[m];
        if (ent < 0) continue;
        int t = ent >> 1;
        float wgt = sW[m];
        float* orow = out + (size_t)t * DIM + n0;
        unsafeAtomicAdd(orow + nj + r,       c0[i] * wgt);
        unsafeAtomicAdd(orow + nj + 16 + r,  c1[i] * wgt);
    }
}

// ---------------------------------------------------------------------------
extern "C" void kernel_launch(void* const* d_in, const int* in_sizes, int n_in,
                              void* d_out, int out_size, void* d_ws, size_t ws_size,
                              hipStream_t stream) {
    const float* x  = (const float*)d_in[0];
    const float* gw = (const float*)d_in[1];
    const float* w1 = (const float*)d_in[2];
    const float* w3 = (const float*)d_in[3];
    const float* w2 = (const float*)d_in[4];
    float* out = (float*)d_out;

    char* ws = (char*)d_ws;
    size_t off = 0;
    unsigned short* xb = (unsigned short*)(ws + off);   off += (size_t)T_TOK * DIM * 2;       // 16 MB
    unsigned short* act = (unsigned short*)(ws + off);  off += (size_t)2 * T_TOK * FDIM * 2;  // 64 MB
    float* topw = (float*)(ws + off);                   off += (size_t)T_TOK * 2 * 4;
    int*   cnt  = (int*)(ws + off);                     off += 256;
    int*   lists = (int*)(ws + off);                    off += (size_t)NEXP * T_TOK * 4;

    (void)hipMemsetAsync(cnt, 0, NEXP * sizeof(int), stream);
    (void)hipMemsetAsync(out, 0, (size_t)T_TOK * DIM * sizeof(float), stream);

    cvt_x_kernel<<<(T_TOK * DIM / 8) / 256, 256, 0, stream>>>(x, xb);
    gate_kernel<<<T_TOK / 8, 256, 0, stream>>>(x, gw, topw, cnt, lists);
    ffn_up_kernel<<<dim3(FDIM / 64, T_TOK / 64, NEXP), 256, 0, stream>>>(xb, w1, w3, cnt, lists, act);
    ffn_down_kernel<<<dim3(DIM / 64, T_TOK / 64, NEXP), 256, 0, stream>>>(act, w2, cnt, lists, topw, out);
}
